// D3Ctta_70420283785636
// MI455X (gfx1250) — matrix operand order repaired and checked by reading, not compiled, not run on hardware
//
#include <hip/hip_runtime.h>

#define N_PTS 100000
#define K_NN  20
#define D_F   96
#define H_DIM 1024
#define C_CLS 7
#define RIDGE 100.0f
#define NCHUNKS (N_PTS / 32)   // 3125
#define HTILES  (H_DIM / 16)   // 64
#define GSPLIT  8              // gram n-split (grid.z)
#define QSPLIT  32             // qacc n-split

typedef __attribute__((ext_vector_type(16))) __bf16 v16bf;
typedef __attribute__((ext_vector_type(8)))  float  v8f;

__device__ __forceinline__ unsigned short f2bf(float f) {
    union { float f; unsigned u; } x; x.f = f;
    unsigned u = x.u;
    unsigned r = (u + 0x7FFFu + ((u >> 16) & 1u)) >> 16;   // RNE
    return (unsigned short)r;
}
__device__ __forceinline__ float bf2f(unsigned short h) {
    union { unsigned u; float f; } x; x.u = ((unsigned)h) << 16;
    return x.f;
}

// ---------------- proto normalize (tiny) ----------------
__global__ void proto_norm_k(const float* __restrict__ proto, float* __restrict__ pn) {
    int c = threadIdx.x;
    if (c < C_CLS) {
        float s = 0.f;
        for (int d = 0; d < D_F; ++d) { float v = proto[c * D_F + d]; s += v * v; }
        float nrm = sqrtf(s); nrm = fmaxf(nrm, 1e-12f);
        float inv = 1.0f / nrm;
        for (int d = 0; d < D_F; ++d) pn[c * D_F + d] = proto[c * D_F + d] * inv;
    }
}

// ---------------- w_rand -> swizzled bf16 B-fragments ----------------
// B tile (kt,nt): lane L holds n = nt*16+(L&15), K = kt*32 + (L>>4)*16 + e
__global__ void wswz_k(const float* __restrict__ w, unsigned short* __restrict__ wswz) {
    int t = blockIdx.x * blockDim.x + threadIdx.x;     // 192 tiles * 32 lanes = 6144
    if (t >= HTILES * 3 * 32) return;
    int lane = t & 31, tile = t >> 5;
    int kt = tile % 3, nt = tile / 3;
    int nn = nt * 16 + (lane & 15);
    int kb = kt * 32 + ((lane >> 4) * 16);
#pragma unroll
    for (int e = 0; e < 16; ++e)
        wswz[(size_t)t * 16 + e] = f2bf(w[(size_t)(kb + e) * H_DIM + nn]);
}

// ---------------- feat -> swizzled bf16 A-fragments ----------------
// A tile (mt,kt): lane L holds m = mt*16+(L&15), K = kt*32 + (L>>4)*16 + e
__global__ void faswz_k(const float* __restrict__ feat, unsigned short* __restrict__ fa) {
    int t = blockIdx.x * blockDim.x + threadIdx.x;     // 6250*3*32 = 600000
    if (t >= (N_PTS / 16) * 3 * 32) return;
    int lane = t & 31, tile = t >> 5;
    int kt = tile % 3, mt = tile / 3;
    int row = mt * 16 + (lane & 15);
    int kb = kt * 32 + ((lane >> 4) * 16);
#pragma unroll
    for (int e = 0; e < 16; ++e) {
        int k = kb + e;
        fa[(size_t)t * 16 + e] = f2bf(feat[(size_t)row * D_F + k]);
    }
}

// ---------------- prototype logits + argmax ----------------
__global__ void preds_k(const float* __restrict__ feat, const float* __restrict__ pn,
                        int* __restrict__ preds) {
    __shared__ float pns[C_CLS * D_F];
    for (int i = threadIdx.x; i < C_CLS * D_F; i += blockDim.x) pns[i] = pn[i];
    __syncthreads();
    int n = blockIdx.x * blockDim.x + threadIdx.x;
    if (n >= N_PTS) return;
    float acc[C_CLS];
#pragma unroll
    for (int c = 0; c < C_CLS; ++c) acc[c] = 0.f;
    for (int d = 0; d < D_F; ++d) {
        float f = feat[(size_t)n * D_F + d];
#pragma unroll
        for (int c = 0; c < C_CLS; ++c) acc[c] += f * pns[c * D_F + d];
    }
    int best = 0; float bv = acc[0];
#pragma unroll
    for (int c = 1; c < C_CLS; ++c) if (acc[c] > bv) { bv = acc[c]; best = c; }
    preds[n] = best;
}

// ---------------- KNN consistency + prior filter -> mask (0/1 float) ----------------
__global__ void mask_k(const int* __restrict__ preds, const int* __restrict__ nidx,
                       const float* __restrict__ coords, const float* __restrict__ nz,
                       float* __restrict__ mask) {
    int n = blockIdx.x * blockDim.x + threadIdx.x;
    if (n >= N_PTS) return;
    int p = preds[n];
    int cnt = 0;
#pragma unroll
    for (int k = 0; k < K_NN; ++k)
        cnt += (preds[nidx[(size_t)n * K_NN + k]] == p) ? 1 : 0;
    bool cmask = cnt > 16;                          // mean > 0.8
    bool plane = (p == 2) | (p == 3) | (p == 4);
    bool man   = (p == 5);
    bool other = (p == 0) | (p == 1) | (p == 6);
    float z = coords[(size_t)n * 3 + 2];
    float nzv = nz[n];
    bool ground = plane && (nzv > 0.9f) && (z < -10.0f);
    bool g = ground | other | man;
    bool m = (man && (nzv < 0.1f)) | other | plane;
    mask[n] = (cmask && g && m) ? 1.0f : 0.0f;
}

// ---------------- feat_h = relu(F @ W)  [WMMA bf16] ----------------
// Also emits the masked feat_h directly in gram A-fragment and B-fragment
// layouts (gfa/gfb) so gram_k needs no LDS transpose gather.
__global__ void gemm1_k(const unsigned short* __restrict__ fa,
                        const unsigned short* __restrict__ wswz,
                        const float* __restrict__ mask,
                        unsigned short* __restrict__ fh,
                        unsigned short* __restrict__ gfa,
                        unsigned short* __restrict__ gfb) {
    int mt = blockIdx.x;
    int wave = threadIdx.x >> 5, lane = threadIdx.x & 31;
    const v16bf* fav = reinterpret_cast<const v16bf*>(fa);
    const v16bf* wv  = reinterpret_cast<const v16bf*>(wswz);
    v16bf a[3];
#pragma unroll
    for (int k = 0; k < 3; ++k) a[k] = fav[(size_t)(mt * 3 + k) * 32 + lane];
    int row0 = mt * 16 + ((lane >> 4) * 8);
    int colL = lane & 15;
    float m8[8];
#pragma unroll
    for (int r = 0; r < 8; ++r) m8[r] = mask[row0 + r];
    int nc = mt >> 1;              // n-chunk of 32 rows
    int half = mt & 1;             // which 16-row half of the chunk
    for (int nt = wave; nt < HTILES; nt += 8) {
        v8f acc = {};
#pragma unroll
        for (int k = 0; k < 3; ++k) {
            v16bf b = wv[(size_t)(nt * 3 + k) * 32 + lane];
            acc = __builtin_amdgcn_wmma_f32_16x16x32_bf16(
                false, a[k], false, b, (short)0, acc, false, false);
        }
        int col = nt * 16 + colL;
        unsigned short mh[8];
#pragma unroll
        for (int r = 0; r < 8; ++r) {
            float v = acc[r];
            v = v > 0.f ? v : 0.f;                  // relu
            fh[(size_t)(row0 + r) * H_DIM + col] = f2bf(v);   // unmasked row-major
            mh[r] = f2bf(v * m8[r]);                // masked, for gram fragments
        }
        uint4 pk;
        pk.x = (unsigned)mh[0] | ((unsigned)mh[1] << 16);
        pk.y = (unsigned)mh[2] | ((unsigned)mh[3] << 16);
        pk.z = (unsigned)mh[4] | ((unsigned)mh[5] << 16);
        pk.w = (unsigned)mh[6] | ((unsigned)mh[7] << 16);
        // gram A-fragment: tile it=nt, chunk nc, lane=lane, elements half*8 + r
        size_t ia = ((size_t)(nt * NCHUNKS + nc) * 32 + lane) * 16 + half * 8;
        *reinterpret_cast<uint4*>(gfa + ia) = pk;
        // gram B-fragment: tile jt=nt, chunk nc, lane = lane,
        // elements half*8 + r
        int lb = lane;
        size_t ibx = ((size_t)(nt * NCHUNKS + nc) * 32 + lb) * 16 + half * 8;
        *reinterpret_cast<uint4*>(gfb + ibx) = pk;
    }
}

// ---------------- Q partial accumulate (deterministic, no atomics) ----------------
// grid (H/256, QSPLIT): each (z, h) pair has a unique writer into qpart.
__global__ void qacc_k(const unsigned short* __restrict__ fh, const float* __restrict__ mask,
                       const int* __restrict__ preds, float* __restrict__ qpart) {
    int h = blockIdx.x * blockDim.x + threadIdx.x;       // [0, 1024)
    int z = blockIdx.y;
    int n0 = z * (N_PTS / QSPLIT);
    int n1 = n0 + (N_PTS / QSPLIT);
    float acc[C_CLS];
#pragma unroll
    for (int c = 0; c < C_CLS; ++c) acc[c] = 0.f;
    for (int n = n0; n < n1; ++n) {
        float mk = mask[n];
        if (mk != 0.f) {
            int p = preds[n];
            float v = bf2f(fh[(size_t)n * H_DIM + h]);
#pragma unroll
            for (int c = 0; c < C_CLS; ++c) acc[c] += (p == c) ? v : 0.f;
        }
    }
#pragma unroll
    for (int c = 0; c < C_CLS; ++c)
        qpart[(size_t)z * H_DIM * C_CLS + h * C_CLS + c] = acc[c];
}

__global__ void reduce_q_k(const float* __restrict__ qpart, const float* __restrict__ qin,
                           float* __restrict__ q) {
    int i = blockIdx.x * blockDim.x + threadIdx.x;
    if (i >= H_DIM * C_CLS) return;
    float s = qin[i];
#pragma unroll
    for (int z = 0; z < QSPLIT; ++z) s += qpart[(size_t)z * H_DIM * C_CLS + i];
    q[i] = s;
}

// ---------------- G = fh_m^T @ fh_m  [WMMA bf16, fragment-direct, n-split] ----------------
// grid (8,8,GSPLIT): 128x128 G tile per (x,y); z slices the n-chunk range and
// writes a private f32 partial. 8 waves/block, each 2 i-tiles x 4 j-tiles.
__global__ void gram_k(const unsigned short* __restrict__ gfa,
                       const unsigned short* __restrict__ gfb,
                       float* __restrict__ gpart) {
    int tid = threadIdx.x;
    int wave = tid >> 5, lane = tid & 31;
    int wi = wave >> 1;                 // 0..3 -> i-tiles {2wi, 2wi+1}
    int wj = wave & 1;                  // 0..1 -> j-tiles {4wj .. 4wj+3}
    int it0 = blockIdx.x * 8 + 2 * wi;
    int jt0 = blockIdx.y * 8 + 4 * wj;
    int z = blockIdx.z;
    int nc0 = (z * NCHUNKS) / GSPLIT;
    int nc1 = ((z + 1) * NCHUNKS) / GSPLIT;
    const v16bf* av = reinterpret_cast<const v16bf*>(gfa);
    const v16bf* bv = reinterpret_cast<const v16bf*>(gfb);

    v8f acc[2][4];
#pragma unroll
    for (int a = 0; a < 2; ++a)
#pragma unroll
        for (int b = 0; b < 4; ++b) { v8f zz = {}; acc[a][b] = zz; }

    size_t abase[2], bbase[4];
#pragma unroll
    for (int ti = 0; ti < 2; ++ti) abase[ti] = (size_t)(it0 + ti) * NCHUNKS * 32 + lane;
#pragma unroll
    for (int tj = 0; tj < 4; ++tj) bbase[tj] = (size_t)(jt0 + tj) * NCHUNKS * 32 + lane;

    for (int nc = nc0; nc < nc1; ++nc) {
        v16bf afr[2], bfr[4];
#pragma unroll
        for (int ti = 0; ti < 2; ++ti) afr[ti] = av[abase[ti] + (size_t)nc * 32];
#pragma unroll
        for (int tj = 0; tj < 4; ++tj) bfr[tj] = bv[bbase[tj] + (size_t)nc * 32];
        // prefetch next chunk (global_prefetch_b8)
        if (nc + 1 < nc1) {
#pragma unroll
            for (int ti = 0; ti < 2; ++ti)
                __builtin_prefetch((const void*)(av + abase[ti] + (size_t)(nc + 1) * 32), 0, 1);
#pragma unroll
            for (int tj = 0; tj < 4; ++tj)
                __builtin_prefetch((const void*)(bv + bbase[tj] + (size_t)(nc + 1) * 32), 0, 1);
        }
#pragma unroll
        for (int ti = 0; ti < 2; ++ti)
#pragma unroll
            for (int tj = 0; tj < 4; ++tj)
                acc[ti][tj] = __builtin_amdgcn_wmma_f32_16x16x32_bf16(
                    false, afr[ti], false, bfr[tj], (short)0, acc[ti][tj], false, false);
    }

    float* gp = gpart + (size_t)z * H_DIM * H_DIM;
#pragma unroll
    for (int ti = 0; ti < 2; ++ti) {
        int row0 = (it0 + ti) * 16 + ((lane >> 4) * 8);
#pragma unroll
        for (int tj = 0; tj < 4; ++tj) {
            int col = (jt0 + tj) * 16 + (lane & 15);
#pragma unroll
            for (int rr = 0; rr < 8; ++rr)
                gp[(size_t)(row0 + rr) * H_DIM + col] = acc[ti][tj][rr];
        }
    }
}

__global__ void reduce_g_k(const float* __restrict__ gpart, const float* __restrict__ gin,
                           float* __restrict__ G) {
    int i = blockIdx.x * blockDim.x + threadIdx.x;
    if (i >= H_DIM * H_DIM) return;
    float s = gin[i];
#pragma unroll
    for (int z = 0; z < GSPLIT; ++z) s += gpart[(size_t)z * H_DIM * H_DIM + i];
    G[i] = s;
}

// ---------------- Cholesky factor + fwd/back solve (single 1024-thread WG) ----------------
__global__ void solve_k(float* __restrict__ G, const float* __restrict__ Q,
                        float* __restrict__ wo) {
    __shared__ float colk[H_DIM];
    __shared__ float Y[H_DIM * C_CLS];      // 28 KB
    int tid = threadIdx.x;
    G[(size_t)tid * H_DIM + tid] += RIDGE;
    for (int i = tid; i < H_DIM * C_CLS; i += blockDim.x) Y[i] = Q[i];
    __syncthreads();

    // in-place right-looking Cholesky (lower triangle)
    for (int k = 0; k < H_DIM; ++k) {
        if (tid == 0) {
            float d = sqrtf(G[(size_t)k * H_DIM + k]);
            G[(size_t)k * H_DIM + k] = d;
            colk[k] = d;
        }
        __syncthreads();
        float dinv = 1.0f / colk[k];
        int i = k + 1 + tid;
        float lik = 0.f;
        if (i < H_DIM) {
            lik = G[(size_t)i * H_DIM + k] * dinv;
            G[(size_t)i * H_DIM + k] = lik;
            colk[i] = lik;
        }
        __syncthreads();
        if (i < H_DIM) {
            for (int j = k + 1; j <= i; ++j)
                G[(size_t)i * H_DIM + j] -= lik * colk[j];
        }
        __syncthreads();
    }

    // forward solve: L Y' = Q
    for (int j = 0; j < H_DIM; ++j) {
        if (tid < C_CLS) Y[j * C_CLS + tid] /= G[(size_t)j * H_DIM + j];
        __syncthreads();
        int i = j + 1 + tid;
        if (i < H_DIM) {
            float lij = G[(size_t)i * H_DIM + j];
#pragma unroll
            for (int c = 0; c < C_CLS; ++c) Y[i * C_CLS + c] -= lij * Y[j * C_CLS + c];
        }
        __syncthreads();
    }
    // backward solve: L^T wo = Y'
    for (int j = H_DIM - 1; j >= 0; --j) {
        if (tid < C_CLS) Y[j * C_CLS + tid] /= G[(size_t)j * H_DIM + j];
        __syncthreads();
        if (tid < j) {
            float lji = G[(size_t)j * H_DIM + tid];
#pragma unroll
            for (int c = 0; c < C_CLS; ++c) Y[tid * C_CLS + c] -= lji * Y[j * C_CLS + c];
        }
        __syncthreads();
    }
    for (int i = tid; i < H_DIM * C_CLS; i += blockDim.x) wo[i] = Y[i];
}

// ---------------- pred_domain = feat_h @ wo  (wave per row) ----------------
__global__ void out_k(const unsigned short* __restrict__ fh, const float* __restrict__ wo,
                      float* __restrict__ out) {
    __shared__ float wos[H_DIM * C_CLS];
    for (int i = threadIdx.x; i < H_DIM * C_CLS; i += blockDim.x) wos[i] = wo[i];
    __syncthreads();
    int wave = threadIdx.x >> 5, lane = threadIdx.x & 31;
    int n = blockIdx.x * 8 + wave;                 // 12500 * 8 = 100000 exactly
    float acc[C_CLS];
#pragma unroll
    for (int c = 0; c < C_CLS; ++c) acc[c] = 0.f;
    const unsigned short* row = fh + (size_t)n * H_DIM;
    for (int h = lane; h < H_DIM; h += 32) {
        float v = bf2f(row[h]);
#pragma unroll
        for (int c = 0; c < C_CLS; ++c) acc[c] += v * wos[h * C_CLS + c];
    }
#pragma unroll
    for (int c = 0; c < C_CLS; ++c) {
        float v = acc[c];
#pragma unroll
        for (int off = 16; off > 0; off >>= 1) v += __shfl_xor(v, off, 32);
        if (lane == 0) out[(size_t)n * C_CLS + c] = v;
    }
}

// ---------------- host side ----------------
static inline size_t align256(size_t x) { return (x + 255) & ~(size_t)255; }

extern "C" void kernel_launch(void* const* d_in, const int* in_sizes, int n_in,
                              void* d_out, int out_size, void* d_ws, size_t ws_size,
                              hipStream_t stream) {
    const float* feat   = (const float*)d_in[0];   // [N, 96]
    const float* proto  = (const float*)d_in[1];   // [7, 96]
    const float* w_rand = (const float*)d_in[2];   // [96, 1024]
    const float* q_in   = (const float*)d_in[3];   // [1024, 7] zeros
    const float* g_in   = (const float*)d_in[4];   // [1024, 1024] zeros
    const float* coords = (const float*)d_in[5];   // [N, 3]
    const float* nz     = (const float*)d_in[6];   // [N]
    const int*   nidx   = (const int*)d_in[7];     // [N, 20] (int32; JAX default)

    char* ws = (char*)d_ws;
    size_t o = 0;
    size_t preds_off = o; o = align256(o + (size_t)N_PTS * 4);
    size_t mask_off  = o; o = align256(o + (size_t)N_PTS * 4);
    size_t pn_off    = o; o = align256(o + (size_t)C_CLS * D_F * 4);
    size_t fa_off    = o; o = align256(o + (size_t)N_PTS * D_F * 2);
    size_t wswz_off  = o; o = align256(o + (size_t)D_F * H_DIM * 2);
    size_t fh_off    = o; o = align256(o + (size_t)N_PTS * H_DIM * 2);
    size_t gfa_off   = o; o = align256(o + (size_t)N_PTS * H_DIM * 2);
    size_t gfb_off   = o; o = align256(o + (size_t)N_PTS * H_DIM * 2);
    size_t gp_off    = o; o = align256(o + (size_t)GSPLIT * H_DIM * H_DIM * 4);
    size_t qp_off    = o; o = align256(o + (size_t)QSPLIT * H_DIM * C_CLS * 4);
    size_t g_off     = o; o = align256(o + (size_t)H_DIM * H_DIM * 4);
    size_t q_off     = o; o = align256(o + (size_t)H_DIM * C_CLS * 4);
    size_t wo_off    = o; o = align256(o + (size_t)H_DIM * C_CLS * 4);

    int*            preds = (int*)(ws + preds_off);
    float*          mask  = (float*)(ws + mask_off);
    float*          pn    = (float*)(ws + pn_off);
    unsigned short* fa    = (unsigned short*)(ws + fa_off);
    unsigned short* wswz  = (unsigned short*)(ws + wswz_off);
    unsigned short* fh    = (unsigned short*)(ws + fh_off);
    unsigned short* gfa   = (unsigned short*)(ws + gfa_off);
    unsigned short* gfb   = (unsigned short*)(ws + gfb_off);
    float*          gpart = (float*)(ws + gp_off);
    float*          qpart = (float*)(ws + qp_off);
    float*          Gw    = (float*)(ws + g_off);
    float*          Qw    = (float*)(ws + q_off);
    float*          wo    = (float*)(ws + wo_off);
    float*          outp  = (float*)d_out;

    proto_norm_k<<<1, 32, 0, stream>>>(proto, pn);
    wswz_k<<<(HTILES * 3 * 32 + 255) / 256, 256, 0, stream>>>(w_rand, wswz);
    faswz_k<<<((N_PTS / 16) * 3 * 32 + 255) / 256, 256, 0, stream>>>(feat, fa);
    preds_k<<<(N_PTS + 255) / 256, 256, 0, stream>>>(feat, pn, preds);
    mask_k<<<(N_PTS + 255) / 256, 256, 0, stream>>>(preds, nidx, coords, nz, mask);
    gemm1_k<<<N_PTS / 16, 256, 0, stream>>>(fa, wswz, mask, fh, gfa, gfb);
    qacc_k<<<dim3(H_DIM / 256, QSPLIT), 256, 0, stream>>>(fh, mask, preds, qpart);
    reduce_q_k<<<(H_DIM * C_CLS + 255) / 256, 256, 0, stream>>>(qpart, q_in, Qw);
    gram_k<<<dim3(HTILES / 8, HTILES / 8, GSPLIT), 256, 0, stream>>>(gfa, gfb, gpart);
    reduce_g_k<<<(H_DIM * H_DIM + 255) / 256, 256, 0, stream>>>(gpart, g_in, Gw);
    solve_k<<<1, 1024, 0, stream>>>(Gw, Qw, wo);
    out_k<<<N_PTS / 8, 256, 0, stream>>>(fh, wo, outp);
}
